// GraphTransformer2DSpatioTemporalModel_30227979829646
// MI455X (gfx1250) — compile-verified, matrix operand order checked by reading, not run on hardware
//
#include <hip/hip_runtime.h>
#include <hip/hip_bf16.h>
#include <math.h>

// ---- model dims ----
#define NB 128
#define S_ 4
#define F_ 8
#define NODES 32
#define T_ 24
#define D_ 128
#define H_ 8
#define HD_ 16
#define DEG 8
#define NN (NB*NODES)          // 4096
#define EE (NN*DEG)            // 32768
#define NT (NN*T_)             // 98304 tokens
#define DFLAT (D_*NODES*T_)    // 98304
static const size_t NTD = (size_t)NT * D_;   // 12,582,912

typedef _Float16 half_t;
typedef __attribute__((ext_vector_type(16))) _Float16 v16h;
typedef __attribute__((ext_vector_type(8)))  _Float16 v8h;
typedef __attribute__((ext_vector_type(8)))  float    v8f;
typedef __attribute__((ext_vector_type(2)))  float    v2f;

// ================= WMMA GEMM (16x64 tile / wave): out[M,N] = A[M,K](f16) * W[N,K]^T(f16) + bias =================
// One A fragment reused across 4 N-subtiles; all B fragments loaded into distinct regs before the WMMA
// sequence so the compiler can stage s_wait_loadcnt instead of fully draining before each WMMA.
__global__ __launch_bounds__(128)
void gemm_f16x4_kernel(const half_t* __restrict__ A, const half_t* __restrict__ W,
                       const float* __restrict__ bias, float* __restrict__ out,
                       half_t* __restrict__ out16, int M, int Nout, int K, int relu)
{
    const int wid = blockIdx.x * 4 + (threadIdx.x >> 5);
    const int tilesN4 = Nout >> 6;
    const int tilesM  = M >> 4;
    if (wid >= tilesM * tilesN4) return;         // wave-uniform
    const int tm  = wid / tilesN4;
    const int tn4 = wid - tm * tilesN4;
    const int lane = threadIdx.x & 31;
    const int hi = lane >> 4;                    // K-half selector
    const int l  = lane & 15;
    const size_t rowA = (size_t)(tm * 16 + l) * (size_t)K;
    const int kb = hi * 8;
    size_t rowW[4];
#pragma unroll
    for (int j = 0; j < 4; ++j) rowW[j] = (size_t)(tn4 * 64 + j * 16 + l) * (size_t)K;
    v8f acc[4] = {v8f{}, v8f{}, v8f{}, v8f{}};
    for (int k0 = 0; k0 < K; k0 += 32) {
        const half_t* pa = A + rowA + k0 + kb;
        __builtin_prefetch(pa + 256, 0, 0);
        v8h a0 = *(const v8h*)pa;
        v8h a1 = *(const v8h*)(pa + 16);
        v8h b0[4], b1[4];
#pragma unroll
        for (int j = 0; j < 4; ++j) {
            const half_t* pw = W + rowW[j] + k0 + kb;
            b0[j] = *(const v8h*)pw;
            b1[j] = *(const v8h*)(pw + 16);
        }
        v16h av;
#pragma unroll
        for (int i = 0; i < 8; ++i) { av[i] = a0[i]; av[i + 8] = a1[i]; }
#pragma unroll
        for (int j = 0; j < 4; ++j) {
            v16h bv;
#pragma unroll
            for (int i = 0; i < 8; ++i) { bv[i] = b0[j][i]; bv[i + 8] = b1[j][i]; }
            acc[j] = __builtin_amdgcn_wmma_f32_16x16x32_f16(false, av, false, bv, (short)0, acc[j], false, false);
        }
    }
#pragma unroll
    for (int j = 0; j < 4; ++j) {
        const int col = tn4 * 64 + j * 16 + l;
        const float bval = bias ? bias[col] : 0.f;
#pragma unroll
        for (int i = 0; i < 8; ++i) {
            const int row = tm * 16 + hi * 8 + i;
            float v = acc[j][i] + bval;
            if (relu) v = v > 0.f ? v : 0.f;
            const size_t o = (size_t)row * Nout + col;
            out[o] = v;
            if (out16) out16[o] = (half_t)v;
        }
    }
}

// ================= WMMA GEMM (16x16 tile / wave) — fallback for narrow N (head1, N=32) =================
__global__ __launch_bounds__(128)
void gemm_f16_kernel(const half_t* __restrict__ A, const half_t* __restrict__ W,
                     const float* __restrict__ bias, float* __restrict__ out,
                     half_t* __restrict__ out16, int M, int Nout, int K, int relu)
{
    const int wid = blockIdx.x * 4 + (threadIdx.x >> 5);
    const int tilesN = Nout >> 4;
    const int tilesM = M >> 4;
    if (wid >= tilesM * tilesN) return;
    const int tm = wid / tilesN;
    const int tn = wid - tm * tilesN;
    const int lane = threadIdx.x & 31;
    const int hi = lane >> 4;
    const int l  = lane & 15;
    const size_t rowA = (size_t)(tm * 16 + l) * (size_t)K;
    const size_t rowW = (size_t)(tn * 16 + l) * (size_t)K;
    const int kb = hi * 8;
    v8f acc = {};
    for (int k0 = 0; k0 < K; k0 += 32) {
        const half_t* pa = A + rowA + k0 + kb;
        const half_t* pw = W + rowW + k0 + kb;
        v8h a0 = *(const v8h*)pa;
        v8h a1 = *(const v8h*)(pa + 16);
        v8h b0 = *(const v8h*)pw;
        v8h b1 = *(const v8h*)(pw + 16);
        v16h av, bv;
#pragma unroll
        for (int i = 0; i < 8; ++i) { av[i] = a0[i]; av[i+8] = a1[i]; bv[i] = b0[i]; bv[i+8] = b1[i]; }
        acc = __builtin_amdgcn_wmma_f32_16x16x32_f16(false, av, false, bv, (short)0, acc, false, false);
    }
    const int col = tn * 16 + l;
    const float bval = bias ? bias[col] : 0.f;
#pragma unroll
    for (int i = 0; i < 8; ++i) {
        const int row = tm * 16 + hi * 8 + i;
        float v = acc[i] + bval;
        if (relu) v = v > 0.f ? v : 0.f;
        const size_t o = (size_t)row * Nout + col;
        out[o] = v;
        if (out16) out16[o] = (half_t)v;
    }
}

// ======= f32 WMMA GEMM (V_WMMA_F32_16X16X4_F32, 16x64/wave) — for the 805MB agg0 weight (HBM-bound) =======
__global__ __launch_bounds__(128)
void gemm_f32x4_kernel(const float* __restrict__ A, const float* __restrict__ W,
                       const float* __restrict__ bias, float* __restrict__ out,
                       half_t* __restrict__ out16, int M, int Nout, int K, int relu)
{
    const int wid = blockIdx.x * 4 + (threadIdx.x >> 5);
    const int tilesN4 = Nout >> 6;
    const int tilesM  = M >> 4;
    if (wid >= tilesM * tilesN4) return;
    const int tm  = wid / tilesN4;
    const int tn4 = wid - tm * tilesN4;
    const int lane = threadIdx.x & 31;
    const int hi = lane >> 4;
    const int l  = lane & 15;
    const size_t rowA = (size_t)(tm * 16 + l) * (size_t)K;
    size_t rowW[4];
#pragma unroll
    for (int j = 0; j < 4; ++j) rowW[j] = (size_t)(tn4 * 64 + j * 16 + l) * (size_t)K;
    v8f acc[4] = {v8f{}, v8f{}, v8f{}, v8f{}};
    for (int k0 = 0; k0 < K; k0 += 4) {
        v2f a = *(const v2f*)(A + rowA + k0 + hi * 2);
        v2f b[4];
#pragma unroll
        for (int j = 0; j < 4; ++j) b[j] = *(const v2f*)(W + rowW[j] + k0 + hi * 2);
#pragma unroll
        for (int j = 0; j < 4; ++j)
            acc[j] = __builtin_amdgcn_wmma_f32_16x16x4_f32(false, a, false, b[j], (short)0, acc[j], false, false);
    }
#pragma unroll
    for (int j = 0; j < 4; ++j) {
        const int col = tn4 * 64 + j * 16 + l;
        const float bval = bias ? bias[col] : 0.f;
#pragma unroll
        for (int i = 0; i < 8; ++i) {
            const int row = tm * 16 + hi * 8 + i;
            float v = acc[j][i] + bval;
            if (relu) v = v > 0.f ? v : 0.f;
            const size_t o = (size_t)row * Nout + col;
            out[o] = v;
            if (out16) out16[o] = (half_t)v;
        }
    }
}

// ================= embedding + 2D positional encoding =================
__global__ __launch_bounds__(128)
void embed_kernel(const float* __restrict__ x, const float* __restrict__ w_in,
                  const float* __restrict__ b_in, float* __restrict__ h32,
                  half_t* __restrict__ h16)
{
    const int nt = blockIdx.x;          // token
    const int d  = threadIdx.x;         // 0..127
    const int n  = nt / T_;
    const int t  = nt - n * T_;
    const int node = n & (NODES - 1);
    int p, j;
    if (d < 64) { p = node; j = d; } else { p = t; j = d - 64; }
    const int i2 = j >> 1;
    const float dv  = __expf(-(float)(2 * i2) * 0.14391156463f);  // ln(1e4)/64
    const float ang = (float)p * dv;
    const float pe  = (j & 1) ? __cosf(ang) : __sinf(ang);
    const float v = x[nt] * w_in[d] + b_in[d] + pe;
    const size_t o = (size_t)nt * D_ + d;
    h32[o] = v;
    h16[o] = (half_t)v;
}

// ================= graph attention: per-edge scores =================
__global__ void gscore_kernel(const float* __restrict__ q, const float* __restrict__ k,
                              const int* __restrict__ src, float* __restrict__ scr)
{
    const int gid = blockIdx.x * blockDim.x + threadIdx.x;   // (e*T + t)*H + h
    if (gid >= EE * T_ * H_) return;
    const int h  = gid & (H_ - 1);
    const int et = gid >> 3;
    const int t  = et % T_;
    const int e  = et / T_;
    const int sN = src[e];
    const int dN = e >> 3;                                   // dst = e / DEG
    const float* qp = q + ((size_t)(dN * T_ + t)) * D_ + h * HD_;
    const float* kp = k + ((size_t)(sN * T_ + t)) * D_ + h * HD_;
    float a = 0.f;
#pragma unroll
    for (int i = 0; i < HD_; ++i) a += qp[i] * kp[i];
    scr[gid] = a * 0.25f;                                    // 1/sqrt(HD)
}

// ================= graph attention: softmax over 8 edges + aggregate + skip =================
__global__ void gagg_kernel(const float* __restrict__ scr, const float* __restrict__ v,
                            const float* __restrict__ sk, const int* __restrict__ src,
                            float* __restrict__ o32, half_t* __restrict__ o16)
{
    const int gid = blockIdx.x * blockDim.x + threadIdx.x;   // (n*T + t)*H + h
    if (gid >= NN * T_ * H_) return;
    const int h  = gid & (H_ - 1);
    const int ntI = gid >> 3;
    const int t  = ntI % T_;
    const int n  = ntI / T_;
    float sc[DEG]; float mx = -1e30f;
#pragma unroll
    for (int j = 0; j < DEG; ++j) {
        sc[j] = scr[((size_t)(n * DEG + j) * T_ + t) * H_ + h];
        mx = fmaxf(mx, sc[j]);
    }
    float den = 0.f;
#pragma unroll
    for (int j = 0; j < DEG; ++j) { sc[j] = __expf(sc[j] - mx); den += sc[j]; }
    const float inv = 1.f / den;
    float accv[HD_];
#pragma unroll
    for (int i = 0; i < HD_; ++i) accv[i] = 0.f;
    for (int j = 0; j < DEG; ++j) {
        const int s = src[n * DEG + j];
        const float* vp = v + ((size_t)(s * T_ + t)) * D_ + h * HD_;
        const float a = sc[j] * inv;
#pragma unroll
        for (int i = 0; i < HD_; ++i) accv[i] += a * vp[i];
    }
    const size_t base = ((size_t)(n * T_ + t)) * D_ + h * HD_;
#pragma unroll
    for (int i = 0; i < HD_; ++i) {
        const float val = accv[i] + sk[base + i];
        o32[base + i] = val;
        o16[base + i] = (half_t)val;
    }
}

// ================= generic small-L MHA (qkv packed [NT,384]) =================
// mode 0: over stations (L=4, stride 192); 1: over features (L=8, stride 24); 2: temporal (L=24, stride 1)
__global__ void mha_kernel(const float* __restrict__ qkv, float* __restrict__ o32,
                           half_t* __restrict__ o16, int mode)
{
    const int gid = blockIdx.x * blockDim.x + threadIdx.x;
    const int nseq = (mode == 0) ? NB * F_ * T_ : (mode == 1) ? NB * S_ * T_ : NN;
    if (gid >= nseq * H_) return;
    const int h   = gid & (H_ - 1);
    const int seq = gid >> 3;
    size_t base; int stride, L;
    if (mode == 0) {
        const int t = seq % T_; const int bf = seq / T_;
        const int f = bf & 7;   const int b = bf >> 3;
        base = (size_t)(b * NODES + f) * T_ + t; stride = F_ * T_; L = S_;
    } else if (mode == 1) {
        const int t = seq % T_; const int bs = seq / T_;
        const int s = bs & 3;   const int b = bs >> 2;
        base = (size_t)(b * NODES + s * F_) * T_ + t; stride = T_; L = F_;
    } else {
        base = (size_t)seq * T_; stride = 1; L = T_;
    }
    const int hoff = h * HD_;
    for (int qi = 0; qi < L; ++qi) {
        const float* qp = qkv + (base + (size_t)qi * stride) * 384 + hoff;
        float scv[T_]; float mx = -1e30f;
        for (int kj = 0; kj < L; ++kj) {
            const float* kp = qkv + (base + (size_t)kj * stride) * 384 + 128 + hoff;
            float a = 0.f;
#pragma unroll
            for (int i = 0; i < HD_; ++i) a += qp[i] * kp[i];
            a *= 0.25f;
            scv[kj] = a; mx = fmaxf(mx, a);
        }
        float den = 0.f;
        for (int kj = 0; kj < L; ++kj) { scv[kj] = __expf(scv[kj] - mx); den += scv[kj]; }
        const float inv = 1.f / den;
        float accv[HD_];
#pragma unroll
        for (int i = 0; i < HD_; ++i) accv[i] = 0.f;
        for (int kj = 0; kj < L; ++kj) {
            const float* vp = qkv + (base + (size_t)kj * stride) * 384 + 256 + hoff;
            const float a = scv[kj] * inv;
#pragma unroll
            for (int i = 0; i < HD_; ++i) accv[i] += a * vp[i];
        }
        const size_t ob = (base + (size_t)qi * stride) * D_ + hoff;
#pragma unroll
        for (int i = 0; i < HD_; ++i) { o32[ob + i] = accv[i]; o16[ob + i] = (half_t)accv[i]; }
    }
}

// ================= residual add + layernorm (one wave32 per token) =================
__global__ __launch_bounds__(128)
void resln_kernel(const float* __restrict__ A, const float* __restrict__ Bv,
                  const float* __restrict__ g, const float* __restrict__ bb,
                  float* __restrict__ o32, half_t* __restrict__ o16)
{
    const int tok  = blockIdx.x * 4 + (threadIdx.x >> 5);
    const int lane = threadIdx.x & 31;
    const size_t base = (size_t)tok * D_ + lane * 4;
    float v[4];
#pragma unroll
    for (int i = 0; i < 4; ++i) v[i] = A[base + i] + Bv[base + i];
    float s = v[0] + v[1] + v[2] + v[3];
#pragma unroll
    for (int m = 16; m >= 1; m >>= 1) s += __shfl_xor(s, m, 32);
    const float mu = s * (1.f / 128.f);
    float qv = 0.f;
#pragma unroll
    for (int i = 0; i < 4; ++i) { const float d = v[i] - mu; qv += d * d; }
#pragma unroll
    for (int m = 16; m >= 1; m >>= 1) qv += __shfl_xor(qv, m, 32);
    const float rstd = rsqrtf(qv * (1.f / 128.f) + 1e-5f);
#pragma unroll
    for (int i = 0; i < 4; ++i) {
        const int d = lane * 4 + i;
        const float o = (v[i] - mu) * rstd * g[d] + bb[d];
        o32[base + i] = o;
        o16[base + i] = (half_t)o;
    }
}

// ================= weight conversions =================
__global__ void cast16_kernel(const float* __restrict__ in, half_t* __restrict__ out, int n)
{
    const int i = blockIdx.x * blockDim.x + threadIdx.x;
    if (i < n) out[i] = (half_t)in[i];
}
// graph W[H,D,HD] -> W'[h*16+e][d]
__global__ void convgw_kernel(const float* __restrict__ in, half_t* __restrict__ out)
{
    const int i = blockIdx.x * blockDim.x + threadIdx.x;   // 16384
    const int nrow = i >> 7; const int d = i & 127;
    const int hh = nrow >> 4; const int e = nrow & 15;
    out[i] = (half_t)in[(hh * D_ + d) * HD_ + e];
}

// ================= final head: sigmoid(x @ W2[1,32]^T + b2) =================
__global__ void headfin_kernel(const float* __restrict__ in, const float* __restrict__ W,
                               const float* __restrict__ b, float* __restrict__ out)
{
    const int i = threadIdx.x;
    if (i < NB) {
        float a = 0.f;
#pragma unroll
        for (int j = 0; j < 32; ++j) a += in[i * 32 + j] * W[j];
        out[i] = 1.f / (1.f + __expf(-(a + b[0])));
    }
}

// =====================================================================================
extern "C" void kernel_launch(void* const* d_in, const int* in_sizes, int n_in,
                              void* d_out, int out_size, void* d_ws, size_t ws_size,
                              hipStream_t stream) {
    (void)in_sizes; (void)n_in; (void)out_size; (void)ws_size;
    // ---- input indices (jax sorted-key flattening of params dict) ----
    // 0:x 1:edge_index 2:batch
    // 3..10: aggW0,aggb0,aggW1,aggb1,aggW2,aggb2,aggW3,aggb3
    // 11: b_in
    // graph L: base 12+12L: +0 Wk +1 Wo +2 Wq +3 Wsk +4 Wv +5 b +6 bk +7 bo +8 bq +9 bsk +10 bv +11 g
    // 36..41: head W0,b0,W1,b1,W2,b2
    // spatial L: base 42+18L: +0 W1 +1 W2 +2 b1 +3 b2 +4 fWo +5 fWqkv +6 fbo +7 fbqkv
    //                          +8 ln1b +9 ln1g +10 ln2b +11 ln2g +12 ln3b +13 ln3g +14 sWo +15 sWqkv +16 sbo +17 sbqkv
    // temporal L: base 78+12L: +0 W1 +1 W2 +2 Wo +3 Wqkv +4 b1 +5 b2 +6 bo +7 bqkv +8 ln1b +9 ln1g +10 ln2b +11 ln2g
    // 102: w_in
    auto F = [&](int i) { return (const float*)d_in[i]; };
    const int* srcIdx = (const int*)d_in[1];      // edge_index row 0

    // ---- workspace carve ----
    char* w = (char*)d_ws;
    float*  h32  = (float*)w;  w += NTD * 4;
    half_t* h16  = (half_t*)w; w += NTD * 2;
    float*  P    = (float*)w;  w += NTD * 4 * 4;          // 4 slots of NT*D f32
    float*  ob   = (float*)w;  w += NTD * 4;
    half_t* ob16 = (half_t*)w; w += NTD * 2;
    float*  scr  = (float*)w;  w += (size_t)EE * T_ * H_ * 4;
    half_t* wz   = (half_t*)w;
    float*  P1 = P + NTD; float* P2 = P + 2 * NTD; float* P3 = P + 3 * NTD;
    half_t* ffh    = (half_t*)P2;                         // f16 FF activations [NT,256]
    half_t* pong16 = (half_t*)P1;                         // small f16 ping-pong (agg/head)

    size_t o = 0;
    half_t *gWq[2], *gWk[2], *gWv[2], *gWsk[2], *gWo[2];
    for (int L = 0; L < 2; ++L) {
        gWq[L] = wz + o; o += 16384; gWk[L] = wz + o; o += 16384;
        gWv[L] = wz + o; o += 16384; gWsk[L] = wz + o; o += 16384;
        gWo[L] = wz + o; o += 16384;
    }
    half_t *sQKV[2], *sWO[2], *fQKV[2], *fWO[2], *sW1[2], *sW2[2];
    for (int L = 0; L < 2; ++L) {
        sQKV[L] = wz + o; o += 49152; sWO[L] = wz + o; o += 16384;
        fQKV[L] = wz + o; o += 49152; fWO[L] = wz + o; o += 16384;
        sW1[L]  = wz + o; o += 32768; sW2[L] = wz + o; o += 32768;
    }
    half_t *tQKV[2], *tWO[2], *tW1[2], *tW2[2];
    for (int L = 0; L < 2; ++L) {
        tQKV[L] = wz + o; o += 49152; tWO[L] = wz + o; o += 16384;
        tW1[L]  = wz + o; o += 32768; tW2[L] = wz + o; o += 32768;
    }
    half_t* aW1 = wz + o; o += 2097152;
    half_t* aW2 = wz + o; o += 524288;
    half_t* aW3 = wz + o; o += 65536;
    half_t* hw0 = wz + o; o += 8192;
    half_t* hw1 = wz + o; o += 2048;

    auto C16 = [&](const float* src, half_t* dst, int n) {
        cast16_kernel<<<(n + 255) / 256, 256, 0, stream>>>(src, dst, n);
    };
    auto GEMM = [&](const half_t* A, const half_t* Wm, const float* bias,
                    float* out, half_t* out16, int M, int Nn, int K, int relu) {
        if ((Nn & 63) == 0) {
            const int waves = (M / 16) * (Nn / 64);
            gemm_f16x4_kernel<<<(waves + 3) / 4, 128, 0, stream>>>(A, Wm, bias, out, out16, M, Nn, K, relu);
        } else {
            const int waves = (M / 16) * (Nn / 16);
            gemm_f16_kernel<<<(waves + 3) / 4, 128, 0, stream>>>(A, Wm, bias, out, out16, M, Nn, K, relu);
        }
    };

    // ---- weight conversions (every call; deterministic) ----
    for (int L = 0; L < 2; ++L) {
        const int GB = 12 + 12 * L;
        convgw_kernel<<<64, 256, 0, stream>>>(F(GB + 2), gWq[L]);
        convgw_kernel<<<64, 256, 0, stream>>>(F(GB + 0), gWk[L]);
        convgw_kernel<<<64, 256, 0, stream>>>(F(GB + 4), gWv[L]);
        convgw_kernel<<<64, 256, 0, stream>>>(F(GB + 3), gWsk[L]);
        C16(F(GB + 1), gWo[L], 16384);
    }
    for (int L = 0; L < 2; ++L) {
        const int SB = 42 + 18 * L;
        C16(F(SB + 15), sQKV[L], 49152); C16(F(SB + 14), sWO[L], 16384);
        C16(F(SB + 5),  fQKV[L], 49152); C16(F(SB + 4),  fWO[L], 16384);
        C16(F(SB + 0),  sW1[L],  32768); C16(F(SB + 1),  sW2[L], 32768);
    }
    for (int L = 0; L < 2; ++L) {
        const int TB = 78 + 12 * L;
        C16(F(TB + 3), tQKV[L], 49152); C16(F(TB + 2), tWO[L], 16384);
        C16(F(TB + 0), tW1[L],  32768); C16(F(TB + 1), tW2[L], 32768);
    }
    C16(F(5), aW1, 2097152); C16(F(7), aW2, 524288); C16(F(9), aW3, 65536);
    C16(F(36), hw0, 8192);   C16(F(38), hw1, 2048);

    // ---- embed + positional encoding ----
    embed_kernel<<<NT, 128, 0, stream>>>(F(0), F(102), F(11), h32, h16);

    // ---- graph transformer layers ----
    for (int L = 0; L < 2; ++L) {
        const int GB = 12 + 12 * L;
        GEMM(h16, gWq[L],  F(GB + 8), P,  nullptr, NT, 128, 128, 0);   // q
        GEMM(h16, gWk[L],  F(GB + 6), P1, nullptr, NT, 128, 128, 0);   // k
        GEMM(h16, gWv[L],  F(GB + 10), P2, nullptr, NT, 128, 128, 0);  // v
        GEMM(h16, gWsk[L], F(GB + 9), P3, nullptr, NT, 128, 128, 0);   // skip
        gscore_kernel<<<(EE * T_ * H_ + 255) / 256, 256, 0, stream>>>(P, P1, srcIdx, scr);
        gagg_kernel<<<(NN * T_ * H_ + 255) / 256, 256, 0, stream>>>(scr, P2, P3, srcIdx, ob, ob16);
        GEMM(ob16, gWo[L], F(GB + 7), P, nullptr, NT, 128, 128, 0);    // Wo
        resln_kernel<<<NT / 4, 128, 0, stream>>>(h32, P, F(GB + 11), F(GB + 5), h32, h16);
    }

    // ---- spatial transformer layers ----
    for (int L = 0; L < 2; ++L) {
        const int SB = 42 + 18 * L;
        // station attention (L=4)
        GEMM(h16, sQKV[L], F(SB + 17), P, nullptr, NT, 384, 128, 0);
        mha_kernel<<<(NB * F_ * T_ * H_ + 255) / 256, 256, 0, stream>>>(P, ob, ob16, 0);
        GEMM(ob16, sWO[L], F(SB + 16), P3, nullptr, NT, 128, 128, 0);
        resln_kernel<<<NT / 4, 128, 0, stream>>>(h32, P3, F(SB + 9), F(SB + 8), h32, h16);
        // feature attention (L=8)
        GEMM(h16, fQKV[L], F(SB + 7), P, nullptr, NT, 384, 128, 0);
        mha_kernel<<<(NB * S_ * T_ * H_ + 255) / 256, 256, 0, stream>>>(P, ob, ob16, 1);
        GEMM(ob16, fWO[L], F(SB + 6), P3, nullptr, NT, 128, 128, 0);
        resln_kernel<<<NT / 4, 128, 0, stream>>>(h32, P3, F(SB + 11), F(SB + 10), h32, h16);
        // feed-forward
        GEMM(h16, sW1[L], F(SB + 2), P, ffh, NT, 256, 128, 1);
        GEMM(ffh, sW2[L], F(SB + 3), P3, nullptr, NT, 128, 256, 0);
        resln_kernel<<<NT / 4, 128, 0, stream>>>(h32, P3, F(SB + 13), F(SB + 12), h32, h16);
    }

    // ---- temporal transformer layers ----
    for (int L = 0; L < 2; ++L) {
        const int TB = 78 + 12 * L;
        GEMM(h16, tQKV[L], F(TB + 7), P, nullptr, NT, 384, 128, 0);
        mha_kernel<<<(NN * H_ + 255) / 256, 256, 0, stream>>>(P, ob, ob16, 2);
        GEMM(ob16, tWO[L], F(TB + 6), P3, nullptr, NT, 128, 128, 0);
        resln_kernel<<<NT / 4, 128, 0, stream>>>(h32, P3, F(TB + 9), F(TB + 8), h32, h16);
        GEMM(h16, tW1[L], F(TB + 4), P, ffh, NT, 256, 128, 1);
        GEMM(ffh, tW2[L], F(TB + 5), P3, nullptr, NT, 128, 256, 0);
        resln_kernel<<<NT / 4, 128, 0, stream>>>(h32, P3, F(TB + 11), F(TB + 10), h32, h16);
    }

    // ---- aggregation MLP: flat[b] = h32[b*DFLAT ...] (contiguous) ----
    {
        const int waves0 = (NB / 16) * (2048 / 64);
        gemm_f32x4_kernel<<<(waves0 + 3) / 4, 128, 0, stream>>>(h32, F(3), F(4), ob, ob16,
                                                                NB, 2048, DFLAT, 1);    // agg0 (fp32 WMMA)
        GEMM(ob16,   aW1, F(6),  P,  pong16, NB, 1024, 2048, 1);                        // agg1
        GEMM(pong16, aW2, F(8),  ob, ob16,   NB, 512, 1024, 1);                         // agg2
        GEMM(ob16,   aW3, F(10), P,  pong16, NB, 128, 512, 1);                          // agg3
        // head
        GEMM(pong16, hw0, F(37), ob, ob16,   NB, 64, 128, 1);                           // head0
        GEMM(ob16,   hw1, F(39), P,  pong16, NB, 32, 64, 1);                            // head1
        headfin_kernel<<<1, 128, 0, stream>>>(P, F(40), F(41), (float*)d_out);          // sigmoid head2
    }
}
